// Head_7791070675824
// MI455X (gfx1250) — matrix operand order repaired and checked by reading, not compiled, not run on hardware
//
#include <hip/hip_runtime.h>
#include <hip/hip_bf16.h>

#define B_  32
#define T_  2048
#define C_  1024
#define HS_ 64

typedef __attribute__((ext_vector_type(16))) __bf16 v16bf;
typedef __attribute__((ext_vector_type(8)))  float  v8f;

__device__ __forceinline__ v8f wmma_bf16(v16bf a, v16bf b, v8f c) {
  // D = A(16x32 bf16) * B(32x16 bf16) + C(16x16 f32)
  return __builtin_amdgcn_wmma_f32_16x16x32_bf16(false, a, false, b,
                                                 (short)0, c, false, false);
}

// ---------------------------------------------------------------------------
// Pack W[C,HS] fp32 -> bf16 WMMA B-fragment tiles (one-shot, 384 KB total).
// Fragment index: mat*128 + kcIdx*4 + nt ; per fragment 32 lanes x 16 bf16.
// ---------------------------------------------------------------------------
__global__ __launch_bounds__(32)
void packW_kernel(const float* __restrict__ Wq,
                  const float* __restrict__ Wk,
                  const float* __restrict__ Wv,
                  __bf16* __restrict__ wp)
{
  const int lane  = threadIdx.x;
  const int frag  = blockIdx.x;            // 0..383
  const int mat   = frag >> 7;
  const int kcIdx = (frag >> 2) & 31;
  const int nt    = frag & 3;
  const float* W  = (mat == 0) ? Wq : (mat == 1) ? Wk : Wv;
  const int kb    = kcIdx * 32 + ((lane < 16) ? 0 : 16);
  const int col   = nt * 16 + (lane & 15);
  __bf16* dst = wp + ((size_t)frag * 32 + lane) * 16;
#pragma unroll
  for (int j = 0; j < 16; ++j)
    dst[j] = (__bf16)W[(size_t)(kb + j) * HS_ + col];
}

// ---------------------------------------------------------------------------
// Projection: one wave reads 16 rows of x ONCE, produces Q, K^T, V 16x64 bf16
// tiles stored pre-swizzled in WMMA fragment layout.
//   qp : A-layout tiles  [b][t/16][hc<2][lane][16]
//   ktp: B-layout tiles  [b][s/16][hc<2][lane][16]   (K-dim = h)
//   vp : B-layout tiles  [b][s/32][ht<4][lane][16]   (K-dim = s)
// ---------------------------------------------------------------------------
__global__ __launch_bounds__(32)
void proj_kernel(const float* __restrict__ x,
                 const __bf16* __restrict__ wp,
                 __bf16* __restrict__ qp,
                 __bf16* __restrict__ ktp,
                 __bf16* __restrict__ vp)
{
  const int lane  = threadIdx.x;
  const int tbase = blockIdx.x * 16;          // row base in flattened [B*T]
  const int mrow  = lane & 15;
  const int kaddA = (lane < 16) ? 0 : 16;
  const int ncol  = lane & 15;
  const int rowoff= (lane < 16) ? 0 : 8;

  v8f acc[12];                                 // [mat][nt]
#pragma unroll
  for (int i = 0; i < 12; ++i) acc[i] = v8f{};

  const float* xrow = x + (size_t)(tbase + mrow) * C_;

  for (int kcIdx = 0; kcIdx < 32; ++kcIdx) {
    const int kc = kcIdx * 32;
    // ---- A fragment: 16x32 chunk of x -> bf16 ------------------------------
    v16bf af;
    const float4* p0 = (const float4*)(xrow + kc + kaddA);
    const float4* p1 = (const float4*)(xrow + kc + 8 + kaddA);
    float4 a0 = p0[0], a1 = p0[1], a2 = p1[0], a3 = p1[1];
    af[0]=(__bf16)a0.x; af[1]=(__bf16)a0.y; af[2]=(__bf16)a0.z; af[3]=(__bf16)a0.w;
    af[4]=(__bf16)a1.x; af[5]=(__bf16)a1.y; af[6]=(__bf16)a1.z; af[7]=(__bf16)a1.w;
    af[8]=(__bf16)a2.x; af[9]=(__bf16)a2.y; af[10]=(__bf16)a2.z; af[11]=(__bf16)a2.w;
    af[12]=(__bf16)a3.x; af[13]=(__bf16)a3.y; af[14]=(__bf16)a3.z; af[15]=(__bf16)a3.w;

    // ---- 12 WMMAs: 3 matrices x 4 N-tiles, B frags contiguous b128 pairs ---
#pragma unroll
    for (int mat = 0; mat < 3; ++mat)
#pragma unroll
      for (int nt = 0; nt < 4; ++nt) {
        const v16bf bf = *(const v16bf*)
            (wp + ((size_t)((mat * 32 + kcIdx) * 4 + nt) * 32 + lane) * 16);
        acc[mat * 4 + nt] = wmma_bf16(af, bf, acc[mat * 4 + nt]);
      }
  }

  // ---- scatter-store into packed fragment layouts --------------------------
  const int b  = tbase >> 11;                  // T_ == 2048
  const int st = (tbase & (T_ - 1)) >> 4;      // 16-row tile index within batch
  const int sc = st >> 1;                      // 32-row chunk index
  const int sshi = (st & 1) * 16;

#pragma unroll
  for (int nt = 0; nt < 4; ++nt) {
#pragma unroll
    for (int r = 0; r < 8; ++r) {
      const int tl = r + rowoff;               // local row 0..15
      // Q: A-layout  (element (tl, h) with h = nt*16+ncol)
      {
        const float valf = acc[0 * 4 + nt][r];
        const int hc  = nt >> 1;
        const int L   = tl + ((nt & 1) ? 16 : 0);
        const int idx = ncol;
        qp[((((size_t)b * 128 + st) * 2 + hc) * 32 + L) * 16 + idx] = (__bf16)valf;
      }
      // K^T: B-layout over (h = K-dim, s = column)
      {
        const float valf = acc[1 * 4 + nt][r];
        const int hc = nt >> 1;
        const int L  = tl + ((nt & 1) ? 16 : 0);
        ktp[((((size_t)b * 128 + st) * 2 + hc) * 32 + L) * 16 + ncol] = (__bf16)valf;
      }
      // V: B-layout over (s = K-dim, h = column)
      {
        const float valf = acc[2 * 4 + nt][r];
        const int L = ncol + sshi;
        vp[((((size_t)b * 64 + sc) * 4 + nt) * 32 + L) * 16 + tl] = (__bf16)valf;
      }
    }
  }
}

// ---------------------------------------------------------------------------
// One causal key chunk (32 keys) of flash attention for a 16-row query block.
// MASKED=true only for the single chunk straddling the diagonal.
// ---------------------------------------------------------------------------
template <bool MASKED>
__device__ __forceinline__
void attn_chunk(int sbase, int qbase, int b, int lane,
                const __bf16* __restrict__ ktp, const __bf16* __restrict__ vp,
                const v16bf (&qf)[2], __bf16* lds,
                v8f (&o)[4], float (&m)[8], float (&l)[8])
{
  const int ncol  = lane & 15;
  const int mrow  = lane & 15;
  const int rowoff= (lane < 16) ? 0 : 8;
  const int kaddA = (lane < 16) ? 0 : 16;
  const float scale = 0.03125f;                // rsqrt(C=1024)
  const float NEGINF = -__builtin_inff();

  const int st0 = sbase >> 4;
  const int scv = sbase >> 5;

  // ---- S = Q * K^T : 16x32 score strip -------------------------------------
  v8f s1 = v8f{}, s2 = v8f{};
#pragma unroll
  for (int hc = 0; hc < 2; ++hc) {
    const v16bf bkA = *(const v16bf*)
        (ktp + ((((size_t)b * 128 + st0) * 2 + hc) * 32 + lane) * 16);
    s1 = wmma_bf16(qf[hc], bkA, s1);
    const v16bf bkB = *(const v16bf*)
        (ktp + ((((size_t)b * 128 + st0 + 1) * 2 + hc) * 32 + lane) * 16);
    s2 = wmma_bf16(qf[hc], bkB, s2);
  }

  // ---- online softmax update -----------------------------------------------
  float p1[8], p2[8], alpha[8];
#pragma unroll
  for (int r = 0; r < 8; ++r) {
    float e1 = s1[r] * scale;
    float e2 = s2[r] * scale;
    if (MASKED) {
      const int qr = qbase + r + rowoff;
      if (sbase + ncol      > qr) e1 = NEGINF;
      if (sbase + 16 + ncol > qr) e2 = NEGINF;
    }
    float mx = fmaxf(e1, e2);
    mx = fmaxf(mx, __shfl_xor(mx, 1));
    mx = fmaxf(mx, __shfl_xor(mx, 2));
    mx = fmaxf(mx, __shfl_xor(mx, 4));
    mx = fmaxf(mx, __shfl_xor(mx, 8));
    const float mnew = fmaxf(m[r], mx);
    const float a = __expf(m[r] - mnew);
    m[r] = mnew;
    alpha[r] = a;
    p1[r] = __expf(e1 - mnew);
    p2[r] = __expf(e2 - mnew);
    float rs = p1[r] + p2[r];
    rs += __shfl_xor(rs, 1);
    rs += __shfl_xor(rs, 2);
    rs += __shfl_xor(rs, 4);
    rs += __shfl_xor(rs, 8);
    l[r] = l[r] * a + rs;
  }
#pragma unroll
  for (int i = 0; i < 4; ++i)
#pragma unroll
    for (int r = 0; r < 8; ++r) o[i][r] *= alpha[r];

  // ---- P: C-layout f32 -> A-layout bf16 via LDS bounce ---------------------
  __syncthreads();
#pragma unroll
  for (int r = 0; r < 8; ++r) {
    const int row = r + rowoff;
    lds[row * 40 + ncol]      = (__bf16)p1[r];
    lds[row * 40 + 16 + ncol] = (__bf16)p2[r];
  }
  __syncthreads();
  v16bf pA;
#pragma unroll
  for (int i = 0; i < 8; ++i) {
    pA[i]     = lds[mrow * 40 + kaddA + i];
    pA[8 + i] = lds[mrow * 40 + 8 + kaddA + i];
  }
  __syncthreads();

  // ---- O += P * V -----------------------------------------------------------
#pragma unroll
  for (int ht = 0; ht < 4; ++ht) {
    const v16bf bv = *(const v16bf*)
        (vp + ((((size_t)b * 64 + scv) * 4 + ht) * 32 + lane) * 16);
    o[ht] = wmma_bf16(pA, bv, o[ht]);
  }
}

// ---------------------------------------------------------------------------
// Flash attention: one wave owns 16 query rows x HS=64. Mask-free main loop
// over fully-causal chunks; single masked chunk on the diagonal.
// ---------------------------------------------------------------------------
__global__ __launch_bounds__(32)
void attn_kernel(const __bf16* __restrict__ qp,
                 const __bf16* __restrict__ ktp,
                 const __bf16* __restrict__ vp,
                 float* __restrict__ out)
{
  __shared__ __bf16 lds[16 * 40];              // padded 16x32 P tile

  const int lane  = threadIdx.x;
  const int qblk  = blockIdx.x;
  const int b     = blockIdx.y;
  const int qbase = qblk * 16;
  const int ncol  = lane & 15;
  const int rowoff= (lane < 16) ? 0 : 8;
  const float NEGINF = -__builtin_inff();

  // ---- Q fragments (A-layout, contiguous) ----------------------------------
  v16bf qf[2];
#pragma unroll
  for (int hc = 0; hc < 2; ++hc)
    qf[hc] = *(const v16bf*)
        (qp + ((((size_t)b * 128 + qblk) * 2 + hc) * 32 + lane) * 16);

  v8f o[4];
#pragma unroll
  for (int i = 0; i < 4; ++i) o[i] = v8f{};
  float m[8], l[8];
#pragma unroll
  for (int r = 0; r < 8; ++r) { m[r] = NEGINF; l[r] = 0.0f; }

  // ---- mask-free main loop: chunks entirely below the diagonal -------------
  int sbase = 0;
  for (; sbase + 32 <= qbase; sbase += 32) {
    // prefetch next chunk's K^T / V fragment tiles
    __builtin_prefetch(ktp + (((size_t)b * 128 + (sbase >> 4) + 2) * 2 * 32) * 16, 0, 3);
    __builtin_prefetch(vp  + (((size_t)b * 64 + (sbase >> 5) + 1) * 4 * 32) * 16, 0, 3);
    attn_chunk<false>(sbase, qbase, b, lane, ktp, vp, qf, lds, o, m, l);
  }
  // ---- single masked chunk straddling the diagonal -------------------------
  attn_chunk<true>(sbase, qbase, b, lane, ktp, vp, qf, lds, o, m, l);

  // ---- normalize + store fp32 ----------------------------------------------
#pragma unroll
  for (int ht = 0; ht < 4; ++ht)
#pragma unroll
    for (int r = 0; r < 8; ++r) {
      const int grow = qbase + r + rowoff;
      out[((size_t)(b * T_ + grow)) * HS_ + ht * 16 + ncol] = o[ht][r] / l[r];
    }
}

extern "C" void kernel_launch(void* const* d_in, const int* in_sizes, int n_in,
                              void* d_out, int out_size, void* d_ws, size_t ws_size,
                              hipStream_t stream) {
  const float* x  = (const float*)d_in[0];
  const float* Wk = (const float*)d_in[1];
  const float* Wq = (const float*)d_in[2];
  const float* Wv = (const float*)d_in[3];
  float* out = (float*)d_out;

  // workspace layout (bf16): packed W frags, then Q / K^T / V fragment tiles
  __bf16* wp  = (__bf16*)d_ws;                         // 384*512      = 196608
  __bf16* qp  = wp  + (size_t)384 * 512;               // B*T*HS       = 4194304
  __bf16* ktp = qp  + (size_t)B_ * T_ * HS_;
  __bf16* vp  = ktp + (size_t)B_ * T_ * HS_;

  dim3 blk(32);
  packW_kernel<<<dim3(384), blk, 0, stream>>>(Wq, Wk, Wv, wp);
  proj_kernel<<<dim3(B_ * T_ / 16), blk, 0, stream>>>(x, wp, qp, ktp, vp);
  attn_kernel<<<dim3(T_ / 16, B_), blk, 0, stream>>>(qp, ktp, vp, out);
}
